// VideoHash_21397527069369
// MI455X (gfx1250) — matrix operand order repaired and checked by reading, not compiled, not run on hardware
//
#include <hip/hip_runtime.h>

typedef _Float16 v16h __attribute__((ext_vector_type(16)));
typedef _Float16 v8h  __attribute__((ext_vector_type(8)));
typedef float    v8f  __attribute__((ext_vector_type(8)));

#define TABLE_SIZE (1u << 19)
#define TMASK (TABLE_SIZE - 1u)
#define PRIME_Y 2654435761u

// floor(16 * 1.3819^l); levels with res <= 723 use dense linear indexing.
__constant__ int kRes[16] = {16, 22, 30, 42, 58, 80, 111, 153,
                             212, 294, 406, 561, 775, 1072, 1481, 2047};

static __device__ __forceinline__ v16h cat16(v8h lo, v8h hh) {
  return __builtin_shufflevector(lo, hh, 0, 1, 2, 3, 4, 5, 6, 7,
                                 8, 9, 10, 11, 12, 13, 14, 15);
}

static __device__ __forceinline__ v8f wmma16(v16h a, v16h b, v8f c) {
  return __builtin_amdgcn_wmma_f32_16x16x32_f16(false, a, false, b,
                                                (short)0, c, false, false);
}

// A-fragment (16x32 f16, M=point, K): lane m holds K 0-15 of ktile,
// lane m+16 holds K 16-31. Two ds_load_b128.
static __device__ __forceinline__ v16h ldA(const _Float16* st, int point,
                                           int hi2, int kt) {
  const v8h* p = (const v8h*)(st + point * 64 + kt * 32 + hi2 * 16);
  return cat16(p[0], p[1]);  // +0..15 halves
}

// B-fragment (32x16 f16, K x N): lane n holds K0-15 of col n, lane n+16 holds
// K16-31 (pre-swizzled in LDS, 32 contiguous bytes per lane).
static __device__ __forceinline__ v16h ldB(const _Float16* w, int tile, int lane) {
  const v8h* p = (const v8h*)(w + tile * 512 + lane * 16);
  return cat16(p[0], p[1]);
}

__global__ __launch_bounds__(256)
void ngp_fused(const float* __restrict__ xs, const float* __restrict__ tables,
               const float* __restrict__ w1, const float* __restrict__ w2,
               const float* __restrict__ w3, float* __restrict__ out,
               int nTiles) {
  __shared__ __align__(16) _Float16 sW1[4096];       // 8 tiles (2 kt x 4 nt)
  __shared__ __align__(16) _Float16 sW2[4096];       // 8 tiles
  __shared__ __align__(16) _Float16 sW3[1024];       // 2 tiles (N padded 3->16)
  __shared__ __align__(16) _Float16 sStage[8][1024]; // per-wave 16 pts x 64 K

  const int tid  = threadIdx.x;
  const int lane = tid & 31;
  const int wv   = tid >> 5;
  const int col  = lane & 15;
  const int hi2  = lane >> 4;

  // ---- stage weights: f32 global -> f16 LDS, in B-fragment layout ----
  {
    const int tile = tid >> 5;               // one 32x16 tile per wave
    const int kt = tile >> 2, nt = tile & 3;
    const int kb = kt * 32 + hi2 * 16;
    const int n  = nt * 16 + col;
#pragma unroll
    for (int h = 0; h < 16; ++h) {
      const int k = kb + h;
      sW1[tile * 512 + lane * 16 + h] =
          (_Float16)((k < 34) ? w1[k * 64 + n] : 0.f);  // pad K 34->64
      sW2[tile * 512 + lane * 16 + h] = (_Float16)w2[k * 64 + n];
    }
    if (tile < 2) {                          // w3: kt = tile, nt = 0
      const int kb3 = tile * 32 + hi2 * 16;
#pragma unroll
      for (int h = 0; h < 16; ++h) {
        const int k = kb3 + h;
        sW3[tile * 512 + lane * 16 + h] =
            (_Float16)((col < 3) ? w3[k * 3 + col] : 0.f);  // pad N 3->16
      }
    }
  }
  __syncthreads();

  _Float16* st = sStage[wv];

  for (int t = blockIdx.x; t < nTiles; t += gridDim.x) {
    const int pbase = t * 128 + wv * 16;

    // zero the staging tile (covers enc padding K 34..63)
    {
      v8h z = {};
      v8h* sp = (v8h*)st;
#pragma unroll
      for (int i = 0; i < 4; ++i) sp[lane * 4 + i] = z;
    }

    // ---- hash-grid encode: lanes 0-15 -> levels 0-7, lanes 16-31 -> 8-15 ----
    {
      const int p = pbase + col;
      const float x = xs[2 * p + 0];
      const float y = xs[2 * p + 1];
      if (hi2 == 0) {
        st[col * 64 + 0] = (_Float16)x;
        st[col * 64 + 1] = (_Float16)y;
      }
      const int lb = hi2 * 8;
#pragma unroll
      for (int li = 0; li < 8; ++li) {
        const int lvl  = lb + li;
        const int resI = kRes[lvl];
        const float res = (float)resI;
        const float px = x * res, py = y * res;
        const float fx = floorf(px), fy = floorf(py);
        const float wx = px - fx, wy = py - fy;
        const unsigned x0 = (unsigned)(int)fx;
        const unsigned y0 = (unsigned)(int)fy;
        const float2* tb =
            (const float2*)(tables + (size_t)lvl * (size_t)(TABLE_SIZE * 2));
        float f0 = 0.f, f1 = 0.f;
#pragma unroll
        for (int c = 0; c < 4; ++c) {
          const unsigned cx = x0 + (c & 1);
          const unsigned cy = y0 + (c >> 1);
          const unsigned dense = cx + cy * (unsigned)(resI + 1);
          const unsigned hsh   = (cx ^ (cy * PRIME_Y)) & TMASK;
          const unsigned idx   = (resI <= 723) ? dense : hsh;
          const float2 f = tb[idx];            // global_load_b64 gather (L2-hot)
          const float wgt =
              ((c & 1) ? wx : 1.f - wx) * ((c >> 1) ? wy : 1.f - wy);
          f0 = fmaf(f.x, wgt, f0);
          f1 = fmaf(f.y, wgt, f1);
        }
        st[col * 64 + 2 + 2 * lvl + 0] = (_Float16)f0;
        st[col * 64 + 2 + 2 * lvl + 1] = (_Float16)f1;
      }
    }

    // ---- layer 1: enc[16x64] @ W1[64x64] ----
    v16h a0 = ldA(st, col, hi2, 0);
    v16h a1 = ldA(st, col, hi2, 1);
    v8f hacc[4];
#pragma unroll
    for (int nt = 0; nt < 4; ++nt) {
      v8f acc = {};
      acc = wmma16(a0, ldB(sW1, nt,     lane), acc);
      acc = wmma16(a1, ldB(sW1, 4 + nt, lane), acc);
      hacc[nt] = acc;
    }
    // ReLU + write h1 back in [point][neuron] f16 (D layout: M=r+8*hi2, N=col)
#pragma unroll
    for (int nt = 0; nt < 4; ++nt)
#pragma unroll
      for (int r = 0; r < 8; ++r)
        st[(r + 8 * hi2) * 64 + nt * 16 + col] =
            (_Float16)fmaxf(hacc[nt][r], 0.f);

    // ---- layer 2: h1[16x64] @ W2[64x64] ----
    a0 = ldA(st, col, hi2, 0);
    a1 = ldA(st, col, hi2, 1);
#pragma unroll
    for (int nt = 0; nt < 4; ++nt) {
      v8f acc = {};
      acc = wmma16(a0, ldB(sW2, nt,     lane), acc);
      acc = wmma16(a1, ldB(sW2, 4 + nt, lane), acc);
      hacc[nt] = acc;
    }
#pragma unroll
    for (int nt = 0; nt < 4; ++nt)
#pragma unroll
      for (int r = 0; r < 8; ++r)
        st[(r + 8 * hi2) * 64 + nt * 16 + col] =
            (_Float16)fmaxf(hacc[nt][r], 0.f);

    // ---- layer 3: h2[16x64] @ W3[64x16(3 valid)] ----
    a0 = ldA(st, col, hi2, 0);
    a1 = ldA(st, col, hi2, 1);
    v8f acc3 = {};
    acc3 = wmma16(a0, ldB(sW3, 0, lane), acc3);
    acc3 = wmma16(a1, ldB(sW3, 1, lane), acc3);

    if (col < 3) {
#pragma unroll
      for (int r = 0; r < 8; ++r)
        out[(size_t)(pbase + r + 8 * hi2) * 3 + col] = acc3[r];
    }
  }
}

extern "C" void kernel_launch(void* const* d_in, const int* in_sizes, int n_in,
                              void* d_out, int out_size, void* d_ws,
                              size_t ws_size, hipStream_t stream) {
  const float* xs     = (const float*)d_in[0];  // [N,2]
  const float* tables = (const float*)d_in[1];  // [16, 2^19, 2]
  const float* w1     = (const float*)d_in[2];  // [34,64]
  const float* w2     = (const float*)d_in[3];  // [64,64]
  const float* w3     = (const float*)d_in[4];  // [64,3]
  float* out          = (float*)d_out;          // [N,3]

  const int n      = in_sizes[0] / 2;           // 2^21
  const int nTiles = n / 128;                   // 128 points per block-iter
  int blocks = nTiles < 1024 ? nTiles : 1024;   // persistent; weights stay in LDS
  if (blocks < 1) blocks = 1;

  ngp_fused<<<blocks, 256, 0, stream>>>(xs, tables, w1, w2, w3, out, nTiles);
}
